// stHGC_77902116815031
// MI455X (gfx1250) — matrix operand order repaired and checked by reading, not compiled, not run on hardware
//
#include <hip/hip_runtime.h>
#include <math.h>

// ---------------- problem constants ----------------
#define N_NODES 8192
#define N_EDGES 262144
#define IN_DIM  1024
#define HID     512
#define OUT_D   64

typedef __bf16 bf16_t;
typedef __attribute__((ext_vector_type(16))) __bf16 v16bf;
typedef __attribute__((ext_vector_type(8)))  __bf16 v8bf;
typedef __attribute__((ext_vector_type(8)))  float  v8f;

// round-to-nearest-even f32 -> bf16 (bit-exact, no reliance on fptrunc codegen)
__device__ __forceinline__ bf16_t f2bf(float f) {
  unsigned int u = __float_as_uint(f);
  u += 0x7FFFu + ((u >> 16) & 1u);
  unsigned short h = (unsigned short)(u >> 16);
  bf16_t r;
  __builtin_memcpy(&r, &h, sizeof(r));
  return r;
}

__device__ __forceinline__ float sigm(float x) { return 1.0f / (1.0f + expf(-x)); }

// ---------------- utility kernels ----------------
__global__ void k_zero(float* __restrict__ p, long n) {
  long i = (long)blockIdx.x * blockDim.x + threadIdx.x;
  if (i < n) p[i] = 0.0f;
}

__global__ void k_f2bf(const float* __restrict__ s, bf16_t* __restrict__ d, long n) {
  long i = (long)blockIdx.x * blockDim.x + threadIdx.x;
  if (i < n) d[i] = f2bf(s[i]);
}

__global__ void k_elu2bf(const float* __restrict__ s, bf16_t* __restrict__ d, long n) {
  long i = (long)blockIdx.x * blockDim.x + threadIdx.x;
  if (i < n) {
    float x = s[i];
    d[i] = f2bf(x > 0.0f ? x : (expf(x) - 1.0f));
  }
}

// src [R,C] f32 -> dst [C,R] bf16
__global__ void k_transpose_bf(const float* __restrict__ s, bf16_t* __restrict__ d,
                               int R, int C) {
  long i = (long)blockIdx.x * blockDim.x + threadIdx.x;
  if (i < (long)R * C) {
    int r = (int)(i / C), c = (int)(i % C);
    d[(long)c * R + r] = f2bf(s[i]);
  }
}

// ---------------- WMMA bf16 GEMM ----------------
// C[M,N] = A[M,K] @ B[K,N], with B passed pre-transposed as Bt[N,K].
// Block = 256 threads (8 waves) -> 128x64 output tile. B panel staged in LDS via
// global_load_async_to_lds_b128 (ASYNCcnt), double buffered; A double-buffered in regs.
// All 8 B fragments of a chunk are loaded into distinct registers and pinned there
// with sched_barrier(0) so the WMMA burst sees only partial s_wait_dscnt waits.
#define KC    64            // K elements per chunk
#define BPAD  72            // padded column stride (bf16 elems): kills ds bank conflicts

__global__ void __launch_bounds__(256)
k_gemm_bf16(const bf16_t* __restrict__ A, const bf16_t* __restrict__ Bt,
            float* __restrict__ C, int K, int lda, int ldbt, int ldc, int mBlocks) {
  __shared__ bf16_t smB[2 * 64 * BPAD];   // double-buffered 64-col x 64-K B panel

  const int tid  = (int)threadIdx.x;
  const int wid  = tid >> 5;
  const int lane = tid & 31;
  const int row  = lane & 15;             // A-row / B-col within a 16x16 tile
  const int hi   = lane >> 4;

  const int mb = (int)blockIdx.x % mBlocks;
  const int nb = (int)blockIdx.x / mBlocks;
  const int m0 = mb * 128 + wid * 16;     // this wave's 16 output rows
  const int n0 = nb * 64;                 // block's 64 output cols

  // Cooperative B staging: thread loads 16B segments of cols c0 and c0+32.
  const int c0  = tid >> 3;
  const int seg = tid & 7;
  const bf16_t* bg0 = Bt + (long)(n0 + c0)      * ldbt + seg * 8;
  const bf16_t* bg1 = Bt + (long)(n0 + c0 + 32) * ldbt + seg * 8;
  const unsigned ldsBase = (unsigned)(size_t)(void*)smB;  // as(3) offset in low 32 bits
  const unsigned l0 = ldsBase + (unsigned)(c0 * BPAD + seg * 8) * 2u;
  const unsigned l1 = ldsBase + (unsigned)((c0 + 32) * BPAD + seg * 8) * 2u;

  // A fragment base (16x32 bf16 layout): lanes 0-15 K {0..7,16..23}, lanes 16-31 {8..15,24..31}
  const bf16_t* aRow = A + (long)(m0 + row) * lda + hi * 8;

  const int nChunks = K >> 6;

  auto issueB = [&](int buf, int kb) {
    unsigned boff = (unsigned)(buf * 64 * BPAD * 2);
    const bf16_t* g0 = bg0 + kb;
    const bf16_t* g1 = bg1 + kb;
    unsigned d0 = l0 + boff, d1 = l1 + boff;
    asm volatile("global_load_async_to_lds_b128 %0, %1, off"
                 :: "v"(d0), "v"(g0) : "memory");
    asm volatile("global_load_async_to_lds_b128 %0, %1, off"
                 :: "v"(d1), "v"(g1) : "memory");
  };

  v8bf aCur[4], aNxt[4];
  auto loadA = [&](v8bf* a, int kb) {
#pragma unroll
    for (int i = 0; i < 4; ++i)
      a[i] = *(const v8bf*)(aRow + kb + i * 16);
  };

  issueB(0, 0);          // prologue: chunk 0 in flight
  loadA(aCur, 0);

  v8f acc[4] = {};

  for (int kc = 0; kc < nChunks; ++kc) {
    asm volatile("s_wait_asynccnt 0x0" ::: "memory");  // my async stores to LDS done
    __syncthreads();                                   // everyone's done; prev reads retired
    const bool more = (kc + 1 < nChunks);
    if (more) {                                        // uniform branch
      issueB((kc + 1) & 1, (kc + 1) << 6);             // prefetch next B panel (async)
      loadA(aNxt, (kc + 1) << 6);                      // prefetch next A frags (regs)
    }
    const bf16_t* sb = smB + (kc & 1) * 64 * BPAD;

    // Load ALL 8 B fragments into distinct regs (16 independent ds_load_b128).
    v16bf bfr[8];
#pragma unroll
    for (int ks = 0; ks < 2; ++ks) {
#pragma unroll
      for (int t = 0; t < 4; ++t) {
        // B frag (32x16, col-striped): lanes 0-15 K {0..7,16..23}, lanes 16-31 K {8..15,24..31}
        const bf16_t* bp = sb + (t * 16 + row) * BPAD + ks * 32 + hi * 8;
        v8bf b0 = *(const v8bf*)(bp);
        v8bf b1 = *(const v8bf*)(bp + 16);
#pragma unroll
        for (int i = 0; i < 8; ++i) {
          bfr[ks * 4 + t][i]     = b0[i];
          bfr[ks * 4 + t][8 + i] = b1[i];
        }
      }
    }

    v16bf af0, af1;
#pragma unroll
    for (int i = 0; i < 8; ++i) {
      af0[i] = aCur[0][i]; af0[8 + i] = aCur[1][i];
      af1[i] = aCur[2][i]; af1[8 + i] = aCur[3][i];
    }

    // Scheduling fence: loads may not sink past this, WMMAs may not hoist above it.
    // Keeps all 8 fragments live in distinct registers -> partial dscnt waits only.
    __builtin_amdgcn_sched_barrier(0);

#pragma unroll
    for (int t = 0; t < 4; ++t)
      acc[t] = __builtin_amdgcn_wmma_f32_16x16x32_bf16(
          false, af0, false, bfr[t], (short)0, acc[t], false, false);
#pragma unroll
    for (int t = 0; t < 4; ++t)
      acc[t] = __builtin_amdgcn_wmma_f32_16x16x32_bf16(
          false, af1, false, bfr[4 + t], (short)0, acc[t], false, false);

    __builtin_amdgcn_sched_barrier(0);

    if (more) {
#pragma unroll
      for (int i = 0; i < 4; ++i) aCur[i] = aNxt[i];
    }
  }

  // C/D layout: lanes 0-15 rows 0-7 (vgpr v -> row v), lanes 16-31 rows 8-15
#pragma unroll
  for (int t = 0; t < 4; ++t)
#pragma unroll
    for (int v = 0; v < 8; ++v)
      C[(long)(m0 + v + hi * 8) * ldc + n0 + t * 16 + row] = acc[t][v];
}

// ---------------- attention node scores: asrc[n]=x[n,:]@att_s, adst[n]=x[n,:]@att_d
__global__ void k_attdot(const float* __restrict__ x, const float* __restrict__ as_,
                         const float* __restrict__ ad_, float* __restrict__ os,
                         float* __restrict__ od, int n, int K) {
  int wave = blockIdx.x * (blockDim.x >> 5) + ((int)threadIdx.x >> 5);
  if (wave >= n) return;
  int lane = (int)threadIdx.x & 31;
  const float* xr = x + (long)wave * K;
  float s = 0.f, d = 0.f;
  for (int k = lane; k < K; k += 32) { float v = xr[k]; s += v * as_[k]; d += v * ad_[k]; }
  for (int off = 16; off > 0; off >>= 1) {
    s += __shfl_xor(s, off, 32);
    d += __shfl_xor(d, off, 32);
  }
  if (lane == 0) { os[wave] = s; od[wave] = d; }
}

// ---------------- edge softmax (segment over dst) ----------------
__global__ void k_edge_score(const int* __restrict__ src, const int* __restrict__ dst,
                             const float* __restrict__ as_, const float* __restrict__ ad_,
                             float* __restrict__ ea, int* __restrict__ mbits, int E) {
  int e = blockIdx.x * blockDim.x + threadIdx.x;
  if (e >= E) return;
  float a = sigm(as_[src[e]] + ad_[dst[e]]);   // in (0,1): int-bit order == float order
  ea[e] = a;
  atomicMax(&mbits[dst[e]], __float_as_int(a));
}

__global__ void k_edge_exp(const int* __restrict__ dst, const float* __restrict__ ea,
                           const float* __restrict__ m, float* __restrict__ ee,
                           float* __restrict__ ssum, int E) {
  int e = blockIdx.x * blockDim.x + threadIdx.x;
  if (e >= E) return;
  float v = expf(ea[e] - m[dst[e]]);
  ee[e] = v;
  atomicAdd(&ssum[dst[e]], v);
}

__global__ void k_edge_w(const int* __restrict__ dst, const float* __restrict__ ee,
                         const float* __restrict__ ssum, float* __restrict__ w, int E) {
  int e = blockIdx.x * blockDim.x + threadIdx.x;
  if (e >= E) return;
  w[e] = ee[e] / (ssum[dst[e]] + 1e-16f);
}

// out[dst,:] += w[e] * xt[src,:]   (one thread per (edge, channel))
__global__ void k_edge_scatter(const int* __restrict__ src, const int* __restrict__ dst,
                               const float* __restrict__ w, const float* __restrict__ xt,
                               float* __restrict__ out, long total, int logC) {
  long i = (long)blockIdx.x * blockDim.x + threadIdx.x;
  if (i >= total) return;
  long e = i >> logC;
  int  c = (int)(i & ((1 << logC) - 1));
  int  s = src[e], d = dst[e];
  atomicAdd(&out[((long)d << logC) + c], xt[((long)s << logC) + c] * w[e]);
}

// ---------------- readout (mask @ emb expressed over edges) ----------------
__global__ void k_deg(const int* __restrict__ src, float* __restrict__ deg, int E) {
  int e = blockIdx.x * blockDim.x + threadIdx.x;
  if (e >= E) return;
  atomicAdd(&deg[src[e]], 1.0f);
}

__global__ void k_r_scatter(const int* __restrict__ src, const int* __restrict__ dst,
                            const float* __restrict__ emb, float* __restrict__ vsum,
                            long total) {
  long i = (long)blockIdx.x * blockDim.x + threadIdx.x;
  if (i >= total) return;
  long e = i >> 6;
  int  c = (int)(i & 63);
  atomicAdd(&vsum[((long)src[e] << 6) + c], emb[((long)dst[e] << 6) + c]);
}

__global__ void k_r_norm(const float* __restrict__ vsum, const float* __restrict__ deg,
                         float* __restrict__ out, int n) {
  int wave = blockIdx.x * (blockDim.x >> 5) + ((int)threadIdx.x >> 5);
  if (wave >= n) return;
  int lane = (int)threadIdx.x & 31;
  float dg = deg[wave];
  float g0 = vsum[(long)wave * 64 + lane] / dg;
  float g1 = vsum[(long)wave * 64 + lane + 32] / dg;
  float ss = g0 * g0 + g1 * g1;
  for (int off = 16; off > 0; off >>= 1) ss += __shfl_xor(ss, off, 32);
  float scale = 1.0f / fmaxf(sqrtf(ss), 1e-12f);
  out[(long)wave * 64 + lane]      = sigm(g0 * scale);
  out[(long)wave * 64 + lane + 32] = sigm(g1 * scale);
}

// out[n,0] = ha[n]·(W c[n]) + b ; out[n,1] = hb[n]·(W c[n]) + b  (one wave per node)
__global__ void k_bilinear(const float* __restrict__ ha, const float* __restrict__ hb,
                           const float* __restrict__ c, const float* __restrict__ W,
                           const float* __restrict__ b, float* __restrict__ out, int n) {
  int wave = blockIdx.x * (blockDim.x >> 5) + ((int)threadIdx.x >> 5);
  if (wave >= n) return;
  int lane = (int)threadIdx.x & 31;
  const float* cr = c + (long)wave * 64;
  float u0 = 0.f, u1 = 0.f;
  for (int j = 0; j < 64; ++j) {
    float cj = cr[j];
    u0 += W[lane * 64 + j] * cj;
    u1 += W[(lane + 32) * 64 + j] * cj;
  }
  float ra = ha[(long)wave * 64 + lane] * u0 + ha[(long)wave * 64 + lane + 32] * u1;
  float rb = hb[(long)wave * 64 + lane] * u0 + hb[(long)wave * 64 + lane + 32] * u1;
  for (int off = 16; off > 0; off >>= 1) {
    ra += __shfl_xor(ra, off, 32);
    rb += __shfl_xor(rb, off, 32);
  }
  if (lane == 0) { out[2 * wave] = ra + b[0]; out[2 * wave + 1] = rb + b[0]; }
}

// ---------------- launcher ----------------
static inline unsigned gb(long n, int t) { return (unsigned)((n + t - 1) / t); }

extern "C" void kernel_launch(void* const* d_in, const int* in_sizes, int n_in,
                              void* d_out, int out_size, void* d_ws, size_t ws_size,
                              hipStream_t stream) {
  (void)in_sizes; (void)n_in; (void)out_size; (void)ws_size;
  const float* features = (const float*)d_in[0];
  const float* feat_a   = (const float*)d_in[1];
  const int*   eidx     = (const int*)d_in[2];
  const float* W1       = (const float*)d_in[3];
  const float* att_src  = (const float*)d_in[4];
  const float* att_dst  = (const float*)d_in[5];
  const float* W2       = (const float*)d_in[6];
  const float* head1    = (const float*)d_in[7];
  const float* disc_w   = (const float*)d_in[8];
  const float* disc_b   = (const float*)d_in[9];
  const int* src = eidx;
  const int* dst = eidx + N_EDGES;

  // output layout: z | h4 | ret | ret_a | emb  (all f32)
  float* out_z    = (float*)d_out;
  float* out_h4   = out_z    + (long)N_NODES * OUT_D;
  float* out_ret  = out_h4   + (long)N_NODES * IN_DIM;
  float* out_reta = out_ret  + (long)N_NODES * 2;
  float* out_emb  = out_reta + (long)N_NODES * 2;

  // bump allocator over workspace (~90 MB with reuse)
  char* ws = (char*)d_ws;
  size_t off = 0;
  auto alloc = [&](size_t bytes) -> void* {
    off = (off + 255) & ~(size_t)255;
    void* p = ws + off;
    off += bytes;
    return p;
  };

  bf16_t* fbf    = (bf16_t*)alloc((long)N_NODES * IN_DIM * 2);
  bf16_t* fabf   = (bf16_t*)alloc((long)N_NODES * IN_DIM * 2);
  bf16_t* w1bf   = (bf16_t*)alloc((long)IN_DIM * HID * 2);   // Bt for h4 GEMM
  bf16_t* w1tbf  = (bf16_t*)alloc((long)HID * IN_DIM * 2);   // Bt for x1/xa1 GEMMs
  bf16_t* w2bf   = (bf16_t*)alloc((long)HID * OUT_D * 2);    // Bt for x3 GEMM
  bf16_t* w2tbf  = (bf16_t*)alloc((long)OUT_D * HID * 2);    // Bt for h2/za_h GEMMs
  bf16_t* h1tbf  = (bf16_t*)alloc((long)OUT_D * OUT_D * 2);  // Bt for z/za GEMMs
  float*  x1     = (float*)alloc((long)N_NODES * HID * 4);   // reused: xa1
  float*  pbuf   = (float*)alloc((long)N_NODES * HID * 4);   // prop output, reused 3x
  bf16_t* hbf    = (bf16_t*)alloc((long)N_NODES * HID * 2);  // h1_bf / h3_bf / pa_bf
  float*  x3     = (float*)alloc((long)N_NODES * HID * 4);
  float*  h2     = (float*)alloc((long)N_NODES * OUT_D * 4); // reused: za_h
  bf16_t* h2bf   = (bf16_t*)alloc((long)N_NODES * OUT_D * 2);
  float*  za     = (float*)alloc((long)N_NODES * OUT_D * 4);
  float*  asrc_b = (float*)alloc(N_NODES * 4);
  float*  adst_b = (float*)alloc(N_NODES * 4);
  float*  aasrc  = (float*)alloc(N_NODES * 4);
  float*  aadst  = (float*)alloc(N_NODES * 4);
  float*  mbuf   = (float*)alloc(N_NODES * 4);
  float*  ssum   = (float*)alloc(N_NODES * 4);
  float*  ea     = (float*)alloc((long)N_EDGES * 4);
  float*  ee     = (float*)alloc((long)N_EDGES * 4);
  float*  vsum   = (float*)alloc((long)N_NODES * OUT_D * 4);
  float*  deg    = (float*)alloc(N_NODES * 4);
  float*  emba   = (float*)alloc((long)N_NODES * OUT_D * 4);

  auto gemm = [&](const bf16_t* A, const bf16_t* Bt, float* Cc, int M, int Ncols, int K,
                  int lda, int ldbt, int ldc) {
    int mBlocks = M >> 7;       // 128 rows per block
    int nBlocks = Ncols >> 6;   // 64 cols per block
    k_gemm_bf16<<<mBlocks * nBlocks, 256, 0, stream>>>(A, Bt, Cc, K, lda, ldbt, ldc, mBlocks);
  };

  auto gat_prop = [&](const float* xt, const float* as_, const float* ad_, float* pout) {
    k_zero<<<gb(N_NODES, 256), 256, 0, stream>>>(mbuf, N_NODES);
    k_zero<<<gb(N_NODES, 256), 256, 0, stream>>>(ssum, N_NODES);
    k_zero<<<gb((long)N_NODES * HID, 256), 256, 0, stream>>>(pout, (long)N_NODES * HID);
    k_edge_score<<<gb(N_EDGES, 256), 256, 0, stream>>>(src, dst, as_, ad_, ea, (int*)mbuf, N_EDGES);
    k_edge_exp<<<gb(N_EDGES, 256), 256, 0, stream>>>(dst, ea, mbuf, ee, ssum, N_EDGES);
    k_edge_w<<<gb(N_EDGES, 256), 256, 0, stream>>>(dst, ee, ssum, ea, N_EDGES);
    long tot = (long)N_EDGES * HID;
    k_edge_scatter<<<gb(tot, 256), 256, 0, stream>>>(src, dst, ea, xt, pout, tot, 9);
  };

  auto readout = [&](const float* embin, float* embout) {
    k_zero<<<gb((long)N_NODES * OUT_D, 256), 256, 0, stream>>>(vsum, (long)N_NODES * OUT_D);
    k_zero<<<gb(N_NODES, 256), 256, 0, stream>>>(deg, N_NODES);
    k_deg<<<gb(N_EDGES, 256), 256, 0, stream>>>(src, deg, N_EDGES);
    long tot = (long)N_EDGES * OUT_D;
    k_r_scatter<<<gb(tot, 256), 256, 0, stream>>>(src, dst, embin, vsum, tot);
    k_r_norm<<<gb((long)N_NODES * 32, 256), 256, 0, stream>>>(vsum, deg, embout, N_NODES);
  };

  // ---- weight / input conversions to bf16 (Bt pre-transposed for WMMA B-fragments)
  k_f2bf<<<gb((long)N_NODES * IN_DIM, 256), 256, 0, stream>>>(features, fbf, (long)N_NODES * IN_DIM);
  k_f2bf<<<gb((long)N_NODES * IN_DIM, 256), 256, 0, stream>>>(feat_a, fabf, (long)N_NODES * IN_DIM);
  k_f2bf<<<gb((long)IN_DIM * HID, 256), 256, 0, stream>>>(W1, w1bf, (long)IN_DIM * HID);
  k_transpose_bf<<<gb((long)IN_DIM * HID, 256), 256, 0, stream>>>(W1, w1tbf, IN_DIM, HID);
  k_f2bf<<<gb((long)HID * OUT_D, 256), 256, 0, stream>>>(W2, w2bf, (long)HID * OUT_D);
  k_transpose_bf<<<gb((long)HID * OUT_D, 256), 256, 0, stream>>>(W2, w2tbf, HID, OUT_D);
  k_transpose_bf<<<gb((long)OUT_D * OUT_D, 256), 256, 0, stream>>>(head1, h1tbf, OUT_D, OUT_D);

  // ---- clean branch
  gemm(fbf, w1tbf, x1, N_NODES, HID, IN_DIM, IN_DIM, IN_DIM, HID);          // x1 = features @ W1
  k_attdot<<<gb((long)N_NODES * 32, 256), 256, 0, stream>>>(x1, att_src, att_dst, asrc_b, adst_b, N_NODES, HID);
  gat_prop(x1, asrc_b, adst_b, pbuf);                                       // conv1 propagate
  k_elu2bf<<<gb((long)N_NODES * HID, 256), 256, 0, stream>>>(pbuf, hbf, (long)N_NODES * HID); // h1
  gemm(hbf, w2tbf, h2, N_NODES, OUT_D, HID, HID, HID, OUT_D);               // h2 = h1 @ W2
  k_f2bf<<<gb((long)N_NODES * OUT_D, 256), 256, 0, stream>>>(h2, h2bf, (long)N_NODES * OUT_D);
  gemm(h2bf, w2bf, x3, N_NODES, HID, OUT_D, OUT_D, OUT_D, HID);             // x3 = h2 @ W2^T
  gat_prop(x3, asrc_b, adst_b, pbuf);                                       // conv3 propagate (tied attn)
  k_elu2bf<<<gb((long)N_NODES * HID, 256), 256, 0, stream>>>(pbuf, hbf, (long)N_NODES * HID); // h3
  gemm(hbf, w1bf, out_h4, N_NODES, IN_DIM, HID, HID, HID, IN_DIM);          // h4 = h3 @ W1^T
  gemm(h2bf, h1tbf, out_z, N_NODES, OUT_D, OUT_D, OUT_D, OUT_D, OUT_D);     // z = h2 @ head1

  // ---- corrupted branch
  gemm(fabf, w1tbf, x1, N_NODES, HID, IN_DIM, IN_DIM, IN_DIM, HID);         // xa1 = feat_a @ W1
  k_attdot<<<gb((long)N_NODES * 32, 256), 256, 0, stream>>>(x1, att_src, att_dst, aasrc, aadst, N_NODES, HID);
  gat_prop(x1, aasrc, aadst, pbuf);
  k_f2bf<<<gb((long)N_NODES * HID, 256), 256, 0, stream>>>(pbuf, hbf, (long)N_NODES * HID);   // no elu
  gemm(hbf, w2tbf, h2, N_NODES, OUT_D, HID, HID, HID, OUT_D);               // za @ W2
  k_f2bf<<<gb((long)N_NODES * OUT_D, 256), 256, 0, stream>>>(h2, h2bf, (long)N_NODES * OUT_D);
  gemm(h2bf, h1tbf, za, N_NODES, OUT_D, OUT_D, OUT_D, OUT_D, OUT_D);        // za = (.. @ W2) @ head1

  // ---- readouts + discriminator
  readout(out_z, out_emb);      // emb   = sigmoid(normalize(mask@z / rowsum))
  readout(za, emba);            // emb_a
  k_bilinear<<<gb((long)N_NODES * 32, 256), 256, 0, stream>>>(out_z, za, out_emb, disc_w, disc_b, out_ret, N_NODES);
  k_bilinear<<<gb((long)N_NODES * 32, 256), 256, 0, stream>>>(za, out_z, emba, disc_w, disc_b, out_reta, N_NODES);
}
